// LigandGNN_67929202754020
// MI455X (gfx1250) — matrix operand order repaired and checked by reading, not compiled, not run on hardware
//
#include <hip/hip_runtime.h>

// ---------------------------------------------------------------------------
// Types for CDNA5 WMMA (wave32, v_wmma_f32_16x16x32_bf16)
// ---------------------------------------------------------------------------
typedef __bf16 bf16;
typedef __attribute__((ext_vector_type(8)))  __bf16 v8bf;
typedef __attribute__((ext_vector_type(16))) __bf16 v16bf;
typedef __attribute__((ext_vector_type(8)))  float  v8f;

#define NN   100000
#define EE   400000
#define GG   2000
#define HH   192
#define INDIM 32

__device__ __forceinline__ float bf2f(bf16 b) {
    unsigned short s = __builtin_bit_cast(unsigned short, b);
    unsigned u = ((unsigned)s) << 16;
    return __builtin_bit_cast(float, u);
}
__device__ __forceinline__ bf16 f2bf(float f) {
    unsigned u = __builtin_bit_cast(unsigned, f);
    u += 0x7FFFu + ((u >> 16) & 1u);          // round-to-nearest-even
    return __builtin_bit_cast(bf16, (unsigned short)(u >> 16));
}
__device__ __forceinline__ v16bf ld16(const bf16* p0, const bf16* p1) {
    v8bf lo = *reinterpret_cast<const v8bf*>(p0);
    v8bf hi = *reinterpret_cast<const v8bf*>(p1);
    return __builtin_shufflevector(lo, hi, 0,1,2,3,4,5,6,7,8,9,10,11,12,13,14,15);
}

// ---------------------------------------------------------------------------
// bf16 WMMA GEMM:  C[M,Ncols] = opt_relu( X[M,K] @ W[K,Ncols] + bias )
// W pre-transposed: Wt[Ncols,K] bf16 -> B fragments are contiguous loads.
// Each wave computes a 32x64 strip (2 row tiles x 4 col tiles):
//   - all loads issued before the 8 WMMAs of a k-step (overlap mem & math)
//   - 4 B fragments reused by 2 A fragments (2x arithmetic intensity on B)
// ---------------------------------------------------------------------------
__global__ __launch_bounds__(128)
void k_gemm(const bf16* __restrict__ X, const bf16* __restrict__ Wt,
            const float* __restrict__ bias,
            float* __restrict__ outF, bf16* __restrict__ outB,
            int M, int K, int Ncols, int numRowTiles, int relu)
{
    const int wave = threadIdx.x >> 5;
    const int unit = blockIdx.x * 4 + wave;
    const int rt0  = unit * 2;
    if (rt0 >= numRowTiles) return;                   // wave-uniform exit
    const bool has1 = (rt0 + 1) < numRowTiles;        // wave-uniform
    const int lane = threadIdx.x & 31;
    const int lo   = lane & 15;                        // row (A) / col (B,C) in tile
    const int kh   = lane >> 4;                        // K-half select
    const int colBase = blockIdx.y * 64;

    // A frag per ISA layout: lane row `lo`; kh=0 -> K +0..7,+16..23 ; kh=1 -> +8..15,+24..31
    const bf16* a0p = X + (size_t)(rt0 * 16 + lo) * K + kh * 8;
    const bf16* a1p = X + (size_t)(rt0 * 16 + 16 + lo) * K + kh * 8;

    v8f acc0[4] = {v8f{}, v8f{}, v8f{}, v8f{}};
    v8f acc1[4] = {v8f{}, v8f{}, v8f{}, v8f{}};

    for (int kk = 0; kk < K; kk += 32) {
        v16bf a0 = ld16(a0p + kk, a0p + kk + 16);
        v16bf a1 = a0;
        if (has1) a1 = ld16(a1p + kk, a1p + kk + 16);

        // B frags: lane holds column colBase+t*16+lo; kh=0 -> K +0..7,+16..23 ; kh=1 -> +8..15,+24..31
        v16bf b[4];
        #pragma unroll
        for (int t = 0; t < 4; ++t) {
            const bf16* bp = Wt + (size_t)(colBase + t * 16 + lo) * K + kk + kh * 8;
            b[t] = ld16(bp, bp + 16);
        }
        #pragma unroll
        for (int t = 0; t < 4; ++t)
            acc0[t] = __builtin_amdgcn_wmma_f32_16x16x32_bf16(
                          false, a0, false, b[t], (short)0, acc0[t], false, false);
        if (has1) {
            #pragma unroll
            for (int t = 0; t < 4; ++t)
                acc1[t] = __builtin_amdgcn_wmma_f32_16x16x32_bf16(
                              false, a1, false, b[t], (short)0, acc1[t], false, false);
        }
    }

    // C/D layout: lanes 0-15: N=lane, M=vgpr r; lanes 16-31: N=lane-16, M=8+r
    const int mofs = kh * 8;
    #pragma unroll
    for (int t = 0; t < 4; ++t) {
        const int col = colBase + t * 16 + lo;
        const float bb = bias ? bias[col] : 0.0f;
        #pragma unroll
        for (int r = 0; r < 8; ++r) {
            const size_t m = (size_t)rt0 * 16 + mofs + r;
            float v = acc0[t][r] + bb;
            if (relu) v = fmaxf(v, 0.0f);
            const size_t o = m * Ncols + col;
            if (outF) outF[o] = v;
            if (outB) outB[o] = f2bf(v);
        }
        if (has1) {
            #pragma unroll
            for (int r = 0; r < 8; ++r) {
                const size_t m = (size_t)(rt0 + 1) * 16 + mofs + r;
                float v = acc1[t][r] + bb;
                if (relu) v = fmaxf(v, 0.0f);
                const size_t o = m * Ncols + col;
                if (outF) outF[o] = v;
                if (outB) outB[o] = f2bf(v);
            }
        }
    }
}

// ---------------------------------------------------------------------------
// Helper kernels
// ---------------------------------------------------------------------------
__global__ void k_f2bf(const float* __restrict__ in, bf16* __restrict__ out, int n) {
    int i = blockIdx.x * blockDim.x + threadIdx.x;
    if (i < n) out[i] = f2bf(in[i]);
}

// W[K,Hc] f32 row-major -> Wt[Hc,K] bf16 (transposed for contiguous B frags)
__global__ void k_wprep(const float* __restrict__ W, bf16* __restrict__ Wt, int K, int Hc) {
    int i = blockIdx.x * blockDim.x + threadIdx.x;
    if (i >= K * Hc) return;
    int k = i / Hc, n = i % Hc;
    Wt[(size_t)n * K + k] = f2bf(W[i]);
}

__global__ void k_zero(float* __restrict__ p, int n) {
    int i = blockIdx.x * blockDim.x + threadIdx.x;
    if (i < n) p[i] = 0.0f;
}

// out[dst] += Y[src] over edges; 48 float4 lanes per edge (H=192)
__global__ void k_scatter(const int* __restrict__ src, const int* __restrict__ dst,
                          const float* __restrict__ Y, float* __restrict__ out, int Ecnt)
{
    size_t idx = (size_t)blockIdx.x * blockDim.x + threadIdx.x;
    if (idx >= (size_t)Ecnt * 48) return;
    int e = (int)(idx / 48), q = (int)(idx % 48);
    int s = src[e], d = dst[e];
    float4 v = reinterpret_cast<const float4*>(Y + (size_t)s * HH)[q];
    float* op = out + (size_t)d * HH + q * 4;
    atomicAdd(op + 0, v.x); atomicAdd(op + 1, v.y);
    atomicAdd(op + 2, v.z); atomicAdd(op + 3, v.w);
}

// per-column sum / sumsq partials (blockDim.x == HH, coalesced across columns)
__global__ void k_colstats(const float* __restrict__ A, float* __restrict__ s,
                           float* __restrict__ sq, int Nrows, int rowsPB)
{
    int c  = threadIdx.x;
    int r0 = blockIdx.x * rowsPB;
    int r1 = r0 + rowsPB; if (r1 > Nrows) r1 = Nrows;
    float sum = 0.0f, ssq = 0.0f;
    for (int r = r0; r < r1; ++r) {
        float v = A[(size_t)r * HH + c];
        sum += v; ssq += v * v;
    }
    atomicAdd(&s[c], sum); atomicAdd(&sq[c], ssq);
}

// batchnorm (training stats, biased var) + ReLU, write bf16 activations
__global__ void k_bn(const float* __restrict__ A, const float* __restrict__ s,
                     const float* __restrict__ sq, const float* __restrict__ gamma,
                     const float* __restrict__ beta, bf16* __restrict__ out, int Nrows)
{
    size_t idx = (size_t)blockIdx.x * blockDim.x + threadIdx.x;
    if (idx >= (size_t)Nrows * HH) return;
    int c = (int)(idx % HH);
    float inv = 1.0f / (float)Nrows;
    float mu  = s[c] * inv;
    float var = fmaxf(sq[c] * inv - mu * mu, 0.0f);
    float sc  = rsqrtf(var + 1e-5f) * gamma[c];
    float v   = (A[idx] - mu) * sc + beta[c];
    out[idx]  = f2bf(fmaxf(v, 0.0f));
}

// global mean-pool accumulation: gsum[batch[n]] += h[n], gcnt[batch[n]] += 1
__global__ void k_pool(const bf16* __restrict__ X, const int* __restrict__ batch,
                       float* __restrict__ gsum, float* __restrict__ gcnt, int Nrows)
{
    size_t idx = (size_t)blockIdx.x * blockDim.x + threadIdx.x;
    if (idx >= (size_t)Nrows * 48) return;
    int n = (int)(idx / 48), q = (int)(idx % 48);
    int b = batch[n];
    const bf16* xp = X + (size_t)n * HH + q * 4;
    float* gp = gsum + (size_t)b * HH + q * 4;
    atomicAdd(gp + 0, bf2f(xp[0])); atomicAdd(gp + 1, bf2f(xp[1]));
    atomicAdd(gp + 2, bf2f(xp[2])); atomicAdd(gp + 3, bf2f(xp[3]));
    if (q == 0) atomicAdd(&gcnt[b], 1.0f);
}

__global__ void k_poolfin(const float* __restrict__ gsum, const float* __restrict__ gcnt,
                          bf16* __restrict__ gb, int Gn)
{
    int i = blockIdx.x * blockDim.x + threadIdx.x;
    if (i >= Gn * HH) return;
    int g = i / HH;
    gb[i] = f2bf(gsum[i] / fmaxf(gcnt[g], 1.0f));
}

__global__ void k_outhead(const float* __restrict__ t2, const float* __restrict__ Wout,
                          const float* __restrict__ bout, float* __restrict__ out, int Gn)
{
    int g = blockIdx.x * blockDim.x + threadIdx.x;
    if (g >= Gn) return;
    float acc = bout[0];
    const float* r = t2 + (size_t)g * HH;
    for (int c = 0; c < HH; ++c) acc += r[c] * Wout[c];
    out[g] = acc;
}

// ---------------------------------------------------------------------------
// Host orchestration
// ---------------------------------------------------------------------------
extern "C" void kernel_launch(void* const* d_in, const int* in_sizes, int n_in,
                              void* d_out, int out_size, void* d_ws, size_t ws_size,
                              hipStream_t stream)
{
    (void)in_sizes; (void)n_in; (void)out_size; (void)ws_size;
    const float* x        = (const float*)d_in[0];
    const int*   edge_src = (const int*)  d_in[1];
    const int*   edge_dst = (const int*)  d_in[2];
    const int*   batch    = (const int*)  d_in[3];
    const float* Wrel0    = (const float*)d_in[4];
    const float* brel0    = (const float*)d_in[5];
    const float* Wroot0   = (const float*)d_in[6];
    const float* Wrel     = (const float*)d_in[7];
    const float* brel     = (const float*)d_in[8];
    const float* Wroot    = (const float*)d_in[9];
    const float* gamma    = (const float*)d_in[10];
    const float* beta     = (const float*)d_in[11];
    const float* Wh1      = (const float*)d_in[12];
    const float* bh1      = (const float*)d_in[13];
    const float* Wh2      = (const float*)d_in[14];
    const float* bh2      = (const float*)d_in[15];
    const float* Wout     = (const float*)d_in[16];
    const float* bout     = (const float*)d_in[17];
    float* out = (float*)d_out;

    // --- carve workspace ---
    char* base = (char*)d_ws; size_t off = 0;
    auto alloc = [&](size_t bytes) {
        void* p = base + off; off += (bytes + 255) & ~(size_t)255; return p;
    };
    bf16*  xb     = (bf16*) alloc((size_t)NN * HH * sizeof(bf16));   // activations (bf16)
    float* ybuf   = (float*)alloc((size_t)NN * HH * sizeof(float));  // x@Wrel + brel
    float* accbuf = (float*)alloc((size_t)NN * HH * sizeof(float));  // x@Wroot + scatter
    bf16*  wtA    = (bf16*) alloc((size_t)HH * HH * sizeof(bf16));   // transposed weights
    bf16*  wtB    = (bf16*) alloc((size_t)HH * HH * sizeof(bf16));
    float* colsum = (float*)alloc(2 * HH * sizeof(float));           // sum | sumsq
    float* colsq  = colsum + HH;
    float* gsum   = (float*)alloc(((size_t)GG * HH + GG) * sizeof(float));
    float* gcnt   = gsum + (size_t)GG * HH;
    bf16*  gb     = (bf16*) alloc((size_t)GG * HH * sizeof(bf16));
    bf16*  t1b    = (bf16*) alloc((size_t)GG * HH * sizeof(bf16));
    float* t2f    = (float*)alloc((size_t)GG * HH * sizeof(float));

    const int TPB = 256;
    const int rowTilesN = NN / 16;                          // 6250
    const int unitsN    = (rowTilesN + 1) / 2;              // 3125 (32 rows / wave)
    const dim3 gemmGridN((unitsN + 3) / 4, HH / 64);
    const int rowTilesG = GG / 16;                          // 125
    const int unitsG    = (rowTilesG + 1) / 2;              // 63
    const dim3 gemmGridG((unitsG + 3) / 4, HH / 64);

    // input features -> bf16
    k_f2bf<<<(NN * INDIM + TPB - 1) / TPB, TPB, 0, stream>>>(x, xb, NN * INDIM);

    for (int l = 0; l < 4; ++l) {
        const int K = (l == 0) ? INDIM : HH;
        const float* Wr = (l == 0) ? Wrel0  : Wrel  + (size_t)(l - 1) * HH * HH;
        const float* br = (l == 0) ? brel0  : brel  + (size_t)(l - 1) * HH;
        const float* Wt = (l == 0) ? Wroot0 : Wroot + (size_t)(l - 1) * HH * HH;

        k_wprep<<<(K * HH + TPB - 1) / TPB, TPB, 0, stream>>>(Wr, wtA, K, HH);
        k_wprep<<<(K * HH + TPB - 1) / TPB, TPB, 0, stream>>>(Wt, wtB, K, HH);

        // Y = h@Wrel + brel ; acc = h@Wroot
        k_gemm<<<gemmGridN, 128, 0, stream>>>(xb, wtA, br, ybuf, nullptr,
                                              NN, K, HH, rowTilesN, 0);
        k_gemm<<<gemmGridN, 128, 0, stream>>>(xb, wtB, nullptr, accbuf, nullptr,
                                              NN, K, HH, rowTilesN, 0);
        // acc[dst] += Y[src]
        k_scatter<<<(int)(((size_t)EE * 48 + TPB - 1) / TPB), TPB, 0, stream>>>(
            edge_src, edge_dst, ybuf, accbuf, EE);

        // batchnorm stats + apply + relu -> bf16 activations (in place over xb)
        k_zero<<<(2 * HH + TPB - 1) / TPB, TPB, 0, stream>>>(colsum, 2 * HH);
        k_colstats<<<(NN + 511) / 512, HH, 0, stream>>>(accbuf, colsum, colsq, NN, 512);
        k_bn<<<(int)(((size_t)NN * HH + TPB - 1) / TPB), TPB, 0, stream>>>(
            accbuf, colsum, colsq, gamma + (size_t)l * HH, beta + (size_t)l * HH, xb, NN);
    }

    // global mean pool
    k_zero<<<(GG * HH + GG + TPB - 1) / TPB, TPB, 0, stream>>>(gsum, GG * HH + GG);
    k_pool<<<(int)(((size_t)NN * 48 + TPB - 1) / TPB), TPB, 0, stream>>>(
        xb, batch, gsum, gcnt, NN);
    k_poolfin<<<(GG * HH + TPB - 1) / TPB, TPB, 0, stream>>>(gsum, gcnt, gb, GG);

    // head MLP: relu(g@Wh1+bh1) @ Wh2 + bh2, then @Wout + bout
    k_wprep<<<(HH * HH + TPB - 1) / TPB, TPB, 0, stream>>>(Wh1, wtA, HH, HH);
    k_gemm<<<gemmGridG, 128, 0, stream>>>(gb, wtA, bh1, nullptr, t1b,
                                          GG, HH, HH, rowTilesG, 1);
    k_wprep<<<(HH * HH + TPB - 1) / TPB, TPB, 0, stream>>>(Wh2, wtB, HH, HH);
    k_gemm<<<gemmGridG, 128, 0, stream>>>(t1b, wtB, bh2, t2f, nullptr,
                                          GG, HH, HH, rowTilesG, 0);
    k_outhead<<<(GG + TPB - 1) / TPB, TPB, 0, stream>>>(t2f, Wout, bout, out, GG);
}
